// GAINAggregator_85899346240
// MI455X (gfx1250) — matrix operand order repaired and checked by reading, not compiled, not run on hardware
//
#include <hip/hip_runtime.h>
#include <hip/hip_bf16.h>

typedef __attribute__((ext_vector_type(16))) _Float16 v16h;
typedef __attribute__((ext_vector_type(8)))  _Float16 v8h;
typedef __attribute__((ext_vector_type(8)))  float    v8f;

#define DCOL 128          // D = T = H = O = 128
#define KNEI 25           // neighbors
#define ALPHA 0.2f
#define BN_EPS 1e-5f

// ---------------------------------------------------------------------------
// Kernel 1: prep (single workgroup, 256 threads)
//   c_self  = W_t @ a_att[:128]
//   c_neigh = W_t @ a_att[128:]
//   WcT[n*128+k] = f16( (W_t @ W1)[k,n] )   (column-major f16 B operand)
//   W2T/W3T      = f16 column-major copies of W2/W3
//   zero 6x128 BN stat accumulators
// ---------------------------------------------------------------------------
__global__ __launch_bounds__(256) void prep_kernel(
    const float* __restrict__ W_t, const float* __restrict__ a_att,
    const float* __restrict__ W1,  const float* __restrict__ W2,
    const float* __restrict__ W3,
    float* __restrict__ c_self, float* __restrict__ c_neigh,
    _Float16* __restrict__ WcT, _Float16* __restrict__ W2T,
    _Float16* __restrict__ W3T, float* __restrict__ stats /*6*128*/)
{
    const int tid = threadIdx.x;
    if (tid < DCOL) {
        float s0 = 0.f, s1 = 0.f;
        #pragma unroll 4
        for (int t = 0; t < DCOL; ++t) {
            float w = W_t[tid * DCOL + t];
            s0 += w * a_att[t];
            s1 += w * a_att[DCOL + t];
        }
        c_self[tid]  = s0;
        c_neigh[tid] = s1;
        #pragma unroll
        for (int i = 0; i < 6; ++i) stats[i * DCOL + tid] = 0.f;
    }
    for (int i = tid; i < DCOL * DCOL; i += 256) {
        const int n = i >> 7, k = i & 127;
        float acc = 0.f;
        #pragma unroll 4
        for (int j = 0; j < DCOL; ++j)
            acc += W_t[k * DCOL + j] * W1[j * DCOL + n];
        WcT[i] = (_Float16)acc;
        W2T[i] = (_Float16)W2[k * DCOL + n];
        W3T[i] = (_Float16)W3[k * DCOL + n];
    }
}

// ---------------------------------------------------------------------------
// Kernel 2: attention + aggregation. One 128-thread WG per row n.
//   score_k = lrelu(self.c_self + neigh_k.c_neigh); attn = softmax_k
//   z = (1+eps)*self + sum_k attn_k * neigh_k     -> f16 row-major
// ---------------------------------------------------------------------------
__global__ __launch_bounds__(128) void attn_agg_kernel(
    const float* __restrict__ self_vecs, const float* __restrict__ neigh_vecs,
    const float* __restrict__ c_self, const float* __restrict__ c_neigh,
    const float* __restrict__ eps_p, _Float16* __restrict__ zA, int Nrows)
{
    const int n = blockIdx.x;
    if (n >= Nrows) return;
    const int t = threadIdx.x;

    __shared__ float sred[128];
    __shared__ float part[KNEI * 132];   // padded stride vs 64 banks
    __shared__ float scores[KNEI + 1];   // [KNEI] = self score

    const float sv = self_vecs[(size_t)n * DCOL + t];
    const float cn = c_neigh[t];

    // self score reduction
    sred[t] = sv * c_self[t];
    __syncthreads();
    #pragma unroll
    for (int s = 64; s > 0; s >>= 1) {
        if (t < s) sred[t] += sred[t + s];
        __syncthreads();
    }
    if (t == 0) scores[KNEI] = sred[0];

    // neighbor column loads (coalesced) + score partials
    float nb[KNEI];
    const float* np = neigh_vecs + ((size_t)n * KNEI) * DCOL + t;
    #pragma unroll
    for (int k = 0; k < KNEI; ++k) {
        nb[k] = np[k * DCOL];
        part[k * 132 + t] = nb[k] * cn;
    }
    __syncthreads();
    if (t < KNEI) {
        float s = 0.f;
        #pragma unroll 4
        for (int i = 0; i < 128; ++i) s += part[t * 132 + i];
        scores[t] = s;
    }
    __syncthreads();

    // softmax (computed redundantly per thread)
    const float ss = scores[KNEI];
    float sc[KNEI], mx = -3.4e38f;
    #pragma unroll
    for (int k = 0; k < KNEI; ++k) {
        float v = ss + scores[k];
        v = v > 0.f ? v : ALPHA * v;       // leaky_relu
        sc[k] = v;
        mx = fmaxf(mx, v);
    }
    float sum = 0.f;
    #pragma unroll
    for (int k = 0; k < KNEI; ++k) { sc[k] = __expf(sc[k] - mx); sum += sc[k]; }
    const float inv = 1.f / sum;

    float z = (1.f + eps_p[0]) * sv;
    #pragma unroll
    for (int k = 0; k < KNEI; ++k) z += (sc[k] * inv) * nb[k];

    zA[(size_t)n * DCOL + t] = (_Float16)z;
}

// ---------------------------------------------------------------------------
// Kernel 3: X = A(f16, Nx128) @ W(128x128 f16 col-major) + bias, plus
// per-column sum / sumsq accumulation for BN. 128 rows per WG, 8 waves,
// each wave: 16 rows x 128 cols via 8 col-tiles x 4 k-steps of WMMA f16.
// ---------------------------------------------------------------------------
#define BPAD 136   // padded LDS column stride (halves)

__global__ __launch_bounds__(256) void gemm_bn_kernel(
    const _Float16* __restrict__ A, const _Float16* __restrict__ WT,
    const float* __restrict__ bias, float* __restrict__ X,
    float* __restrict__ csum, float* __restrict__ cssq, int Nrows)
{
    __shared__ _Float16 Bw[DCOL * BPAD];
    __shared__ float sbias[DCOL];
    __shared__ float ssum[DCOL];
    __shared__ float sssq[DCOL];

    const int tid = threadIdx.x;

    // stage W (column-major f16) into padded LDS with 16B vector loads
    const uint4* wsrc = (const uint4*)WT;             // 2048 x 16B
    for (int i = tid; i < 2048; i += 256) {
        const int c  = i >> 4;
        const int k8 = (i & 15) << 3;
        *(uint4*)&Bw[c * BPAD + k8] = wsrc[i];
    }
    if (tid < DCOL) { sbias[tid] = bias[tid]; ssum[tid] = 0.f; sssq[tid] = 0.f; }
    __syncthreads();

    const int wave = tid >> 5;
    const int lane = tid & 31;
    const int l15  = lane & 15;
    const int rowbase = blockIdx.x * 128 + wave * 16;

    int r = rowbase + l15;
    if (r >= Nrows) r = Nrows - 1;                    // clamp (tail tile)
    const _Float16* arow = A + (size_t)r * DCOL;

    const int akoff = (lane < 16) ? 0 : 8;            // A frag: K {0..7,16..23} / {8..15,24..31}
    const int bkoff = (lane < 16) ? 0 : 8;            // B frag: K {0..7,16..23} / {8..15,24..31}

    v8f acc[8];
    #pragma unroll
    for (int i = 0; i < 8; ++i) acc[i] = (v8f)(0.f);

    #pragma unroll
    for (int kk = 0; kk < 4; ++kk) {
        const v8h a0 = *(const v8h*)(arow + kk * 32 + akoff);
        const v8h a1 = *(const v8h*)(arow + kk * 32 + 16 + akoff);
        const v16h af = __builtin_shufflevector(a0, a1,
            0,1,2,3,4,5,6,7,8,9,10,11,12,13,14,15);
        #pragma unroll
        for (int nt = 0; nt < 8; ++nt) {
            const _Float16* bp = &Bw[(nt * 16 + l15) * BPAD + kk * 32 + bkoff];
            const v8h b0 = *(const v8h*)bp;
            const v8h b1 = *(const v8h*)(bp + 16);
            const v16h bf = __builtin_shufflevector(b0, b1,
                0,1,2,3,4,5,6,7,8,9,10,11,12,13,14,15);
            acc[nt] = __builtin_amdgcn_wmma_f32_16x16x32_f16(
                false, af, false, bf, (short)0, acc[nt], false, false);
        }
    }

    // epilogue: bias add, store f32, per-column sum / sumsq
    const int rlo = rowbase + ((lane < 16) ? 0 : 8);
    #pragma unroll
    for (int nt = 0; nt < 8; ++nt) {
        const int col = nt * 16 + l15;
        const float bcol = sbias[col];
        float s = 0.f, q = 0.f;
        #pragma unroll
        for (int j = 0; j < 8; ++j) {
            const int row = rlo + j;
            const float v = acc[nt][j] + bcol;
            if (row < Nrows) {
                X[(size_t)row * DCOL + col] = v;
                s += v;
                q += v * v;
            }
        }
        atomicAdd(&ssum[col], s);
        atomicAdd(&sssq[col], q);
    }
    __syncthreads();
    if (tid < DCOL) {
        atomicAdd(&csum[tid], ssum[tid]);
        atomicAdd(&cssq[tid], sssq[tid]);
    }
}

// ---------------------------------------------------------------------------
// Kernel 4: BN finalize + apply + leaky_relu.  y = lrelu(g*(x-m)/sqrt(v+eps)+b)
// ---------------------------------------------------------------------------
template <bool OUT_F16>
__global__ __launch_bounds__(256) void bn_apply_kernel(
    const float* __restrict__ X, const float* __restrict__ csum,
    const float* __restrict__ cssq, const float* __restrict__ gamma,
    const float* __restrict__ beta, void* __restrict__ out, int Nrows)
{
    __shared__ float ssc[DCOL];
    __shared__ float ssh[DCOL];
    const int tid = threadIdx.x;
    if (tid < DCOL) {
        const float invN = 1.f / (float)Nrows;
        const float m = csum[tid] * invN;
        const float v = cssq[tid] * invN - m * m;
        const float inv = rsqrtf(v + BN_EPS);
        const float sc = gamma[tid] * inv;
        ssc[tid] = sc;
        ssh[tid] = beta[tid] - m * sc;
    }
    __syncthreads();
    const size_t idx = (size_t)blockIdx.x * 256 + tid;
    const size_t total = (size_t)Nrows * DCOL;
    if (idx < total) {
        const int c = (int)(idx & 127);
        float v = ssc[c] * X[idx] + ssh[c];
        v = v > 0.f ? v : ALPHA * v;
        if (OUT_F16) ((_Float16*)out)[idx] = (_Float16)v;
        else         ((float*)out)[idx]    = v;
    }
}

// ---------------------------------------------------------------------------
extern "C" void kernel_launch(void* const* d_in, const int* in_sizes, int n_in,
                              void* d_out, int out_size, void* d_ws, size_t ws_size,
                              hipStream_t stream)
{
    const float* self_vecs = (const float*)d_in[0];
    const float* neigh     = (const float*)d_in[1];
    const float* W_t       = (const float*)d_in[2];
    const float* a_att     = (const float*)d_in[3];
    const float* eps_p     = (const float*)d_in[4];
    const float* W1        = (const float*)d_in[5];
    const float* b1        = (const float*)d_in[6];
    const float* g1        = (const float*)d_in[7];
    const float* be1       = (const float*)d_in[8];
    const float* W2        = (const float*)d_in[9];
    const float* b2        = (const float*)d_in[10];
    const float* g2        = (const float*)d_in[11];
    const float* be2       = (const float*)d_in[12];
    const float* W3        = (const float*)d_in[13];
    const float* b3        = (const float*)d_in[14];
    const float* g3        = (const float*)d_in[15];
    const float* be3       = (const float*)d_in[16];

    const int N = in_sizes[0] / DCOL;   // 50000

    // workspace layout
    char* base = (char*)d_ws;
    float*    c_self  = (float*)(base);                  // 128 f32
    float*    c_neigh = (float*)(base + 512);            // 128 f32
    float*    stats   = (float*)(base + 1024);           // 6*128 f32
    _Float16* WcT     = (_Float16*)(base + 4096);        // 128*128 f16
    _Float16* W2T     = (_Float16*)(base + 4096 + 32768);
    _Float16* W3T     = (_Float16*)(base + 4096 + 65536);
    size_t off = 4096 + 3 * 32768;                       // = 102400
    _Float16* A0 = (_Float16*)(base + off); off += (size_t)N * DCOL * 2;
    _Float16* A1 = (_Float16*)(base + off); off += (size_t)N * DCOL * 2;
    float*    X  = (float*)(base + off);

    float* sum1 = stats + 0 * DCOL; float* ssq1 = stats + 1 * DCOL;
    float* sum2 = stats + 2 * DCOL; float* ssq2 = stats + 3 * DCOL;
    float* sum3 = stats + 4 * DCOL; float* ssq3 = stats + 5 * DCOL;

    prep_kernel<<<1, 256, 0, stream>>>(W_t, a_att, W1, W2, W3,
                                       c_self, c_neigh, WcT, W2T, W3T, stats);

    attn_agg_kernel<<<N, 128, 0, stream>>>(self_vecs, neigh, c_self, c_neigh,
                                           eps_p, A0, N);

    const int gblocks = (N + 127) / 128;
    const int eblocks = (int)(((size_t)N * DCOL + 255) / 256);

    // layer 1 (W_t folded into Wc = W_t @ W1)
    gemm_bn_kernel<<<gblocks, 256, 0, stream>>>(A0, WcT, b1, X, sum1, ssq1, N);
    bn_apply_kernel<true ><<<eblocks, 256, 0, stream>>>(X, sum1, ssq1, g1, be1, A1, N);
    // layer 2
    gemm_bn_kernel<<<gblocks, 256, 0, stream>>>(A1, W2T, b2, X, sum2, ssq2, N);
    bn_apply_kernel<true ><<<eblocks, 256, 0, stream>>>(X, sum2, ssq2, g2, be2, A0, N);
    // layer 3 -> f32 output
    gemm_bn_kernel<<<gblocks, 256, 0, stream>>>(A0, W3T, b3, X, sum3, ssq3, N);
    bn_apply_kernel<false><<<eblocks, 256, 0, stream>>>(X, sum3, ssq3, g3, be3, d_out, N);
}
